// Segmenter_5944234738187
// MI455X (gfx1250) — matrix operand order repaired and checked by reading, not compiled, not run on hardware
//
#include <hip/hip_runtime.h>
#include <hip/hip_bf16.h>
#include <stdint.h>

typedef __attribute__((ext_vector_type(16))) _Float16 v16h;
typedef __attribute__((ext_vector_type(8)))  float    v8f;

#define THREADS 512
#define WAVES   (THREADS / 32)     // 16 waves -> perfect 16x16 WMMA inter-wave scan
#define TPT     32                 // tokens per thread, packed into 32-bit masks
#define SEGWORDS 512               // bitset for up to 16384 segment ids

// ---------------------------------------------------------------------------
// Block-wide exclusive scan over 512 per-thread counts.
// Level 1: intra-wave shuffle scan (wave32).
// Level 2: the 16 wave totals are scanned with ONE v_wmma_f32_16x16x32_f16:
//   D = A x B, A[m][k] = (k < m) ? 1 : 0 (strict lower triangular, f16),
//   B[k][j] = total_k (replicated across columns; rows k>=16 zeroed via A).
//   => D[m][j] = exclusive prefix of wave m, exact (totals <= 1024 < 2048).
// NOTE: wtot/wbase are plain (non-volatile) pointers into __shared__ arrays so
// the compiler promotes accesses to ds_load/ds_store; __syncthreads() provides
// all required ordering. (volatile forced SCOPE_SYS flat stores + full waits.)
// ---------------------------------------------------------------------------
__device__ __forceinline__ int block_exscan(int cnt, int tid, int lane, int w,
                                            int* wtot, int* wbase)
{
    int incl = cnt;
    #pragma unroll
    for (int off = 1; off < 32; off <<= 1) {
        int y = __shfl_up(incl, off, 32);
        if (lane >= off) incl += y;
    }
    if (lane == 31) wtot[w] = incl;          // wave total
    __syncthreads();

    if (tid < 32) {                          // wave 0, EXEC all ones (WMMA req.)
        const int m    = lane & 15;
        const int half = lane >> 4;
        float xv = (lane < 16) ? (float)wtot[lane] : 0.0f;
        _Float16 xh = (_Float16)xv;
        v16h a, bmat;
        #pragma unroll
        for (int e = 0; e < 16; ++e) {
            int r = e >> 1, bit = e & 1;
            // 16-bit A-matrix 16x32 K layout (ISA 7.12.2)
            int K = (r < 4) ? (2 * r + bit + 8 * half)
                            : (16 + 2 * (r - 4) + bit + 8 * half);
            a[e]    = (K < m) ? (_Float16)1.0f : (_Float16)0.0f; // strict => exclusive
            bmat[e] = (K < 16) ? (_Float16)(float)wtot[K] : (_Float16)0.0f; // B row K = total_K
        }
        v8f c = {};
        c = __builtin_amdgcn_wmma_f32_16x16x32_f16(
                /*neg_a=*/false, a, /*neg_b=*/false, bmat,
                /*c_mod=*/(short)0, c, /*reuse_a=*/false, /*reuse_b=*/false);
        // D layout: lane<16 -> M=r (VGPR r); lane>=16 -> M=r+8
        if (lane == 0) {
            #pragma unroll
            for (int r = 0; r < 8; ++r) wbase[r] = (int)(c[r] + 0.5f);
        }
        if (lane == 16) {
            #pragma unroll
            for (int r = 0; r < 8; ++r) wbase[8 + r] = (int)(c[r] + 0.5f);
        }
    }
    __syncthreads();
    return wbase[w] + (incl - cnt);          // block-wide exclusive prefix
}

__global__ __launch_bounds__(THREADS)
void seg_prune_kernel(const int* __restrict__ input_ids,
                      const int* __restrict__ attention_mask,
                      const int* __restrict__ token2page,
                      const unsigned char* __restrict__ keep_pages,
                      const int* __restrict__ query_pos,
                      int* __restrict__ ids_out,
                      int* __restrict__ attn_out,
                      int* __restrict__ pos_out,
                      int L, int P)
{
    __shared__ __align__(16) unsigned char pk_lds[256];  // keep_pages row
    __shared__ unsigned int segbits[SEGWORDS];           // sentence "any kept" bitset
    __shared__ int wtot[WAVES], wbase[WAVES];
    __shared__ int red_any, red_lv;

    const int tid  = threadIdx.x;
    const int lane = tid & 31;
    const int w    = tid >> 5;
    const int b    = blockIdx.x;
    const long long base = (long long)b * L;
    const int i0 = tid * TPT;

    for (int s = tid; s < SEGWORDS; s += THREADS) segbits[s] = 0u;
    if (tid == 0) { red_any = 0; red_lv = -1; }

    // --- CDNA5 async global -> LDS stage of the keep_pages row -------------
    if (tid * 16 < P) {
        unsigned int loff = (unsigned int)(unsigned long long)(void*)pk_lds
                          + (unsigned int)(tid * 16);
        const unsigned char* gp = keep_pages + (long long)b * P + tid * 16;
        asm volatile("global_load_async_to_lds_b128 %0, %1, off"
                     :: "v"(loff), "v"(gp) : "memory");
    }
    asm volatile("s_wait_asynccnt 0" ::: "memory");

    int qp = query_pos[b];
    qp = qp < 0 ? 0 : (qp > L - 1 ? L - 1 : qp);

    __syncthreads();

    // --- phase 1: per-token valid/page-keep/tail mask + boundary bits ------
    unsigned int mbits = 0u, bbits = 0u, vbits = 0u;
    int lv_local = -1;
    #pragma unroll 4
    for (int k = 0; k < TPT; ++k) {
        const int i  = i0 + k;
        const int id = input_ids[base + i];
        const int am = attention_mask[base + i];
        int page     = token2page[base + i];
        page = page < 0 ? 0 : (page > P - 1 ? P - 1 : page);
        const bool valid = (am != 0);
        const bool pk    = (pk_lds[page] != 0);
        bool m   = (valid && pk) || (valid && (i >= qp));
        bool bnd = valid && ((id == 13) || (id == 30));
        mbits |= ((unsigned int)m)     << k;
        bbits |= ((unsigned int)bnd)   << k;
        vbits |= ((unsigned int)valid) << k;
        if (valid) lv_local = i;
    }

    // --- exclusive cumsum of boundary indicators => segment id per token ---
    const int bexcl = block_exscan(__popc(bbits), tid, lane, w, wtot, wbase);

    // --- sentence rescue: segment-max(mask) over context via LDS bitset ----
    for (int k = 0; k < TPT; ++k) {
        const int i = i0 + k;
        if (i < qp && ((mbits >> k) & 1u)) {
            const int seg = bexcl + __popc(bbits & ((1u << k) - 1u));
            atomicOr(&segbits[seg >> 5], 1u << (seg & 31));
        }
    }
    __syncthreads();
    for (int k = 0; k < TPT; ++k) {
        const int i = i0 + k;
        if (i < qp && !((mbits >> k) & 1u) && ((vbits >> k) & 1u)) {
            const int seg = bexcl + __popc(bbits & ((1u << k) - 1u));
            if ((segbits[seg >> 5] >> (seg & 31)) & 1u) mbits |= 1u << k;
        }
    }

    // --- fallback: if nothing kept, keep last valid token (or position 0) --
    if (mbits)          atomicOr(&red_any, 1);
    if (lv_local >= 0)  atomicMax(&red_lv, lv_local);
    __syncthreads();
    if (red_any == 0) {
        const int lv = (red_lv < 0) ? 0 : red_lv;
        mbits = (lv >= i0 && lv < i0 + TPT) ? (1u << (lv - i0)) : 0u;
    }
    __syncthreads();

    // --- stream compaction: exclusive scan of mask -> scatter ranks --------
    const int mexcl = block_exscan(__popc(mbits), tid, lane, w, wtot, wbase);
    const int lk    = wbase[WAVES - 1] + wtot[WAVES - 1];

    for (int k = 0; k < TPT; ++k) {
        const int j  = i0 + k;
        const bool on = (j < lk);
        attn_out[base + j] = on ? 1 : 0;
        pos_out [base + j] = on ? j : 0;
        if (!on) ids_out[base + j] = 0;                   // PAD tail
        if ((mbits >> k) & 1u) {                          // scatter kept id
            const int dst = mexcl + __popc(mbits & ((1u << k) - 1u));
            ids_out[base + dst] = input_ids[base + j];
        }
    }
}

extern "C" void kernel_launch(void* const* d_in, const int* in_sizes, int n_in,
                              void* d_out, int out_size, void* d_ws, size_t ws_size,
                              hipStream_t stream)
{
    const int*           input_ids      = (const int*)d_in[0];
    const int*           attention_mask = (const int*)d_in[1];
    const int*           token2page     = (const int*)d_in[2];
    const unsigned char* keep_pages     = (const unsigned char*)d_in[3]; // jnp bool
    const int*           query_pos      = (const int*)d_in[4];

    const int B = in_sizes[4];
    const int L = in_sizes[0] / B;   // 16384 -> THREADS * TPT
    const int P = in_sizes[3] / B;   // 256

    int* ids_out  = (int*)d_out;
    int* attn_out = ids_out  + (long long)B * L;
    int* pos_out  = attn_out + (long long)B * L;

    seg_prune_kernel<<<B, THREADS, 0, stream>>>(
        input_ids, attention_mask, token2page, keep_pages, query_pos,
        ids_out, attn_out, pos_out, L, P);
}
